// QuantLlamaDecoderLayer_62251255988370
// MI455X (gfx1250) — 16-bit and 8-bit matrix operand order repaired and checked by reading, compiled, not run on hardware
//
#include <hip/hip_runtime.h>
#include <cstdint>

typedef __attribute__((ext_vector_type(8)))  int          v8i;
typedef __attribute__((ext_vector_type(4)))  int          v4i;
typedef __attribute__((ext_vector_type(4)))  unsigned int v4u;
typedef __attribute__((ext_vector_type(8)))  float        v8f;
typedef __attribute__((ext_vector_type(16))) __bf16       v16bf;

#define DEV static __device__ __forceinline__

#if __has_builtin(__builtin_amdgcn_tensor_load_to_lds)
#define USE_TDM 1
#else
#define USE_TDM 0
#endif

typedef __attribute__((address_space(3))) int8_t lds_i8;

constexpr int D_     = 4096;
constexpr int H_     = 32;
constexpr int KVH_   = 8;
constexpr int HD_    = 128;
constexpr int I_     = 14336;
constexpr int BSZ_   = 4;
constexpr int S_     = 1024;
constexpr int T_     = BSZ_ * S_;
constexpr int QDIM_  = H_ * HD_;
constexpr int KVDIM_ = KVH_ * HD_;
constexpr int QKV_   = QDIM_ + 2 * KVDIM_;
constexpr float SOFTCAP_ = 50.0f;
constexpr float EPS_     = 1e-6f;

DEV float wave_max32(float v) {
  #pragma unroll
  for (int o = 16; o >= 1; o >>= 1) v = fmaxf(v, __shfl_xor(v, o, 32));
  return v;
}
DEV float wave_sum32(float v) {
  #pragma unroll
  for (int o = 16; o >= 1; o >>= 1) v += __shfl_xor(v, o, 32);
  return v;
}
DEV float wave_max16(float v) {
  #pragma unroll
  for (int o = 1; o < 16; o <<= 1) v = fmaxf(v, __shfl_xor(v, o, 32));
  return v;
}
DEV float wave_sum16(float v) {
  #pragma unroll
  for (int o = 1; o < 16; o <<= 1) v += __shfl_xor(v, o, 32);
  return v;
}

__global__ __launch_bounds__(256) void rmsnorm_quant_kernel(
    const float* __restrict__ x, const float* __restrict__ w,
    int8_t* __restrict__ q, float* __restrict__ sOut, int n)
{
  const int t = blockIdx.x;
  const float* row = x + (size_t)t * n;
  __shared__ float red[8];
  float ss = 0.f;
  for (int i = threadIdx.x; i < n; i += 256) { float v = row[i]; ss = fmaf(v, v, ss); }
  ss = wave_sum32(ss);
  if ((threadIdx.x & 31) == 0) red[threadIdx.x >> 5] = ss;
  __syncthreads();
  float tot = 0.f;
  #pragma unroll
  for (int i = 0; i < 8; ++i) tot += red[i];
  const float rstd = rsqrtf(tot / (float)n + EPS_);
  __syncthreads();
  float amax = 0.f;
  for (int i = threadIdx.x; i < n; i += 256)
    amax = fmaxf(amax, fabsf(row[i] * rstd * w[i]));
  amax = wave_max32(amax);
  if ((threadIdx.x & 31) == 0) red[threadIdx.x >> 5] = amax;
  __syncthreads();
  float am = 0.f;
  #pragma unroll
  for (int i = 0; i < 8; ++i) am = fmaxf(am, red[i]);
  const float scale = fmaxf(am * (1.f / 127.f), 1e-8f);
  const float inv = 1.f / scale;
  if (threadIdx.x == 0) sOut[t] = scale;
  int8_t* qr = q + (size_t)t * n;
  for (int i = threadIdx.x; i < n; i += 256) {
    float v = rintf(row[i] * rstd * w[i] * inv);
    qr[i] = (int8_t)fminf(fmaxf(v, -128.f), 127.f);
  }
}

__global__ __launch_bounds__(256) void quant_rows_kernel(
    const float* __restrict__ x, int8_t* __restrict__ q,
    float* __restrict__ sOut, int n)
{
  const int t = blockIdx.x;
  const float* row = x + (size_t)t * n;
  __shared__ float red[8];
  float amax = 0.f;
  for (int i = threadIdx.x; i < n; i += 256) amax = fmaxf(amax, fabsf(row[i]));
  amax = wave_max32(amax);
  if ((threadIdx.x & 31) == 0) red[threadIdx.x >> 5] = amax;
  __syncthreads();
  float am = 0.f;
  #pragma unroll
  for (int i = 0; i < 8; ++i) am = fmaxf(am, red[i]);
  const float scale = fmaxf(am * (1.f / 127.f), 1e-8f);
  const float inv = 1.f / scale;
  if (threadIdx.x == 0) sOut[t] = scale;
  int8_t* qr = q + (size_t)t * n;
  for (int i = threadIdx.x; i < n; i += 256) {
    float v = rintf(row[i] * inv);
    qr[i] = (int8_t)fminf(fmaxf(v, -128.f), 127.f);
  }
}

__global__ __launch_bounds__(256) void silu_quant_kernel(
    const float* __restrict__ gu, int8_t* __restrict__ q, float* __restrict__ sOut)
{
  const int t = blockIdx.x;
  const float* g = gu + (size_t)t * (2 * I_);
  const float* u = g + I_;
  __shared__ float red[8];
  float amax = 0.f;
  for (int i = threadIdx.x; i < I_; i += 256) {
    const float gv = g[i];
    const float a = (gv / (1.f + __expf(-gv))) * u[i];
    amax = fmaxf(amax, fabsf(a));
  }
  amax = wave_max32(amax);
  if ((threadIdx.x & 31) == 0) red[threadIdx.x >> 5] = amax;
  __syncthreads();
  float am = 0.f;
  #pragma unroll
  for (int i = 0; i < 8; ++i) am = fmaxf(am, red[i]);
  const float scale = fmaxf(am * (1.f / 127.f), 1e-8f);
  const float inv = 1.f / scale;
  if (threadIdx.x == 0) sOut[t] = scale;
  int8_t* qr = q + (size_t)t * I_;
  for (int i = threadIdx.x; i < I_; i += 256) {
    const float gv = g[i];
    const float a = (gv / (1.f + __expf(-gv))) * u[i];
    float v = rintf(a * inv);
    qr[i] = (int8_t)fminf(fmaxf(v, -128.f), 127.f);
  }
}

__global__ __launch_bounds__(256) void gemm_i8_kernel(
    const int8_t* __restrict__ qx, const float* __restrict__ sx,
    const int8_t* __restrict__ qw, const float* __restrict__ sw,
    const float* __restrict__ residual, float* __restrict__ out,
    int M, int N, int K)
{
  __shared__ alignas(16) int8_t As[2][128 * 64];
  const int tid   = threadIdx.x;
  const int lane  = tid & 31;
  const int wid   = tid >> 5;
  const int wm    = wid >> 2, wn = wid & 3;
  const int laneN = lane & 15;
  const int hi    = lane >> 4;
  const int m0    = blockIdx.y * 128;
  const int nBase = blockIdx.x * 64 + wn * 16;
  const size_t bRow = (size_t)(nBase + laneN) * (size_t)K;
  const int steps = K >> 6;

  v8i acc[4];
  #pragma unroll
  for (int i = 0; i < 4; ++i)
    #pragma unroll
    for (int r = 0; r < 8; ++r) acc[i][r] = 0;

#if USE_TDM
  v8i g1;
  g1[0] = 0;
  g1[1] = (K & 0xFFFF) << 16;
  g1[2] = (int)(((unsigned)K >> 16) | (((unsigned)M & 0xFFFF) << 16));
  g1[3] = (int)(((unsigned)M >> 16) | (64u << 16));
  g1[4] = 128;
  g1[5] = K;
  g1[6] = 0;
  g1[7] = 0;
  v4i gz;
  gz[0] = 0; gz[1] = 0; gz[2] = 0; gz[3] = 0;
  v8i gz8;
  #pragma unroll
  for (int i = 0; i < 8; ++i) gz8[i] = 0;
  const unsigned lds0 = (unsigned)(size_t)(lds_i8*)&As[0][0];
  const unsigned lds1 = (unsigned)(size_t)(lds_i8*)&As[1][0];
  if (wid == 0) {
    const unsigned long long ga0 =
        (unsigned long long)(size_t)(qx + (size_t)m0 * K);
    v4u g0;
    g0[0] = 1u;
    g0[1] = lds0;
    g0[2] = (unsigned)(ga0 & 0xFFFFFFFFull);
    g0[3] = (unsigned)((ga0 >> 32) & 0x1FFFFFFull) | 0x80000000u;
    __builtin_amdgcn_tensor_load_to_lds(g0, g1, gz, gz, gz8, 0);
  }
#endif

  for (int s = 0; s < steps; ++s) {
    const int k0 = s << 6;
#if USE_TDM
    if (wid == 0) {
      if (s + 1 < steps) {
        const unsigned long long ga =
            (unsigned long long)(size_t)(qx + (size_t)m0 * K + (k0 + 64));
        v4u g0;
        g0[0] = 1u;
        g0[1] = ((s + 1) & 1) ? lds1 : lds0;
        g0[2] = (unsigned)(ga & 0xFFFFFFFFull);
        g0[3] = (unsigned)((ga >> 32) & 0x1FFFFFFull) | 0x80000000u;
        __builtin_amdgcn_tensor_load_to_lds(g0, g1, gz, gz, gz8, 0);
        __builtin_amdgcn_s_wait_tensorcnt(1);
      } else {
        __builtin_amdgcn_s_wait_tensorcnt(0);
      }
    }
#else
    {
      const int r = tid >> 1;
      const int c = (tid & 1) * 32;
      const int8_t* src = &qx[(size_t)(m0 + r) * K + k0 + c];
      *reinterpret_cast<int4*>(&As[s & 1][r * 64 + c]) =
          *reinterpret_cast<const int4*>(src);
      *reinterpret_cast<int4*>(&As[s & 1][r * 64 + c + 16]) =
          *reinterpret_cast<const int4*>(src + 16);
    }
#endif
    __syncthreads();
    union { v8i v; int4 d4[2]; } B;
    {
      const int8_t* bp = qw + bRow + k0 + hi * 16;
      B.d4[0] = *reinterpret_cast<const int4*>(bp);
      B.d4[1] = *reinterpret_cast<const int4*>(bp + 32);
      if (k0 + 64 < K) __builtin_prefetch(bp + 64, 0, 1);
    }
    const int8_t* Ab = &As[s & 1][0];
    #pragma unroll
    for (int i = 0; i < 4; ++i) {
      union { v8i v; int2 d2[4]; } A;
      const int8_t* ap = &Ab[(wm * 64 + i * 16 + laneN) * 64 + hi * 16];
      A.d2[0] = *reinterpret_cast<const int2*>(ap + 0);
      A.d2[1] = *reinterpret_cast<const int2*>(ap + 8);
      A.d2[2] = *reinterpret_cast<const int2*>(ap + 32);
      A.d2[3] = *reinterpret_cast<const int2*>(ap + 40);
      acc[i] = __builtin_amdgcn_wmma_i32_16x16x64_iu8(true, A.v, true, B.v, acc[i], false, false);
    }
    __syncthreads();
  }
  const int n = nBase + laneN;
  const float swn = sw[n];
  #pragma unroll
  for (int i = 0; i < 4; ++i) {
    const int mB = m0 + wm * 64 + i * 16;
    #pragma unroll
    for (int r = 0; r < 8; ++r) {
      const int m = mB + r + hi * 8;
      float v = (float)acc[i][r] * sx[m] * swn;
      const size_t idx = (size_t)m * N + n;
      if (residual) v += residual[idx];
      out[idx] = v;
    }
  }
}

__global__ __launch_bounds__(256) void rope_split_kernel(
    const float* __restrict__ qkv, const float* __restrict__ freqs,
    float* __restrict__ qT, float* __restrict__ kT, float* __restrict__ vT)
{
  const int t = blockIdx.x;
  const int b = t / S_, s = t % S_;
  const float* row = qkv + (size_t)t * QKV_;
  const float* fr  = freqs + (size_t)s * (HD_ / 2);
  for (int idx = threadIdx.x; idx < QKV_; idx += 256) {
    const float v = row[idx];
    if (idx < QDIM_) {
      const int h = idx >> 7, d = idx & 127;
      float si, co; __sincosf(fr[d & 63], &si, &co);
      const float o = (d < 64) ? (v * co - row[idx + 64] * si)
                               : (v * co + row[idx - 64] * si);
      qT[(((size_t)b * H_ + h) * S_ + s) * HD_ + d] = o;
    } else if (idx < QDIM_ + KVDIM_) {
      const int j = idx - QDIM_;
      const int h = j >> 7, d = j & 127;
      float si, co; __sincosf(fr[d & 63], &si, &co);
      const float o = (d < 64) ? (v * co - row[idx + 64] * si)
                               : (v * co + row[idx - 64] * si);
      kT[(((size_t)b * KVH_ + h) * S_ + s) * HD_ + d] = o;
    } else {
      const int j = idx - QDIM_ - KVDIM_;
      const int h = j >> 7, d = j & 127;
      vT[(((size_t)b * KVH_ + h) * S_ + s) * HD_ + d] = v;
    }
  }
}

__global__ __launch_bounds__(256) void attention_kernel(
    const float* __restrict__ qT, const float* __restrict__ kT,
    const float* __restrict__ vT, float* __restrict__ attn)
{
  __shared__ alignas(16) __bf16 Ksh[32][128];
  __shared__ alignas(16) __bf16 Vsh[128][32];
  __shared__ alignas(16) __bf16 Psh[8][16][32];
  const int tid   = threadIdx.x;
  const int lane  = tid & 31;
  const int wid   = tid >> 5;
  const int laneN = lane & 15;
  const int hi    = lane >> 4;
  const int b = blockIdx.z, h = blockIdx.y;
  const int q0  = (blockIdx.x * 8 + wid) * 16;
  const int kvh = h >> 2;
  const float qscale = 0.08838834764831845f;

  v16bf qa[4];
  {
    const float* qp = qT + ((((size_t)b * H_ + h) * S_) + q0 + laneN) * HD_;
    #pragma unroll
    for (int c = 0; c < 4; ++c) {
      const int base = c * 32 + hi * 8;
      #pragma unroll
      for (int j = 0; j < 8; ++j) {
        qa[c][j]     = (__bf16)(qp[base + j] * qscale);
        qa[c][j + 8] = (__bf16)(qp[base + 16 + j] * qscale);
      }
    }
  }
  v8f accv[8];
  #pragma unroll
  for (int ch = 0; ch < 8; ++ch)
    #pragma unroll
    for (int r = 0; r < 8; ++r) accv[ch][r] = 0.f;
  float mrow[8], lrow[8];
  #pragma unroll
  for (int r = 0; r < 8; ++r) { mrow[r] = -1e30f; lrow[r] = 0.f; }

  const float* kbase = kT + (((size_t)b * KVH_ + kvh) * S_) * HD_;
  const float* vbase = vT + (((size_t)b * KVH_ + kvh) * S_) * HD_;
  const int kend = (blockIdx.x + 1) * 128;

  for (int j0 = 0; j0 < kend; j0 += 32) {
    for (int e = tid; e < 32 * 128; e += 256) {
      const int kk = e >> 7, dd = e & 127;
      Ksh[kk][dd] = (__bf16)kbase[(size_t)(j0 + kk) * HD_ + dd];
      Vsh[dd][kk] = (__bf16)vbase[(size_t)(j0 + kk) * HD_ + dd];
    }
    __syncthreads();
    if (j0 <= q0 + 15) {
      v8f s0, s1;
      #pragma unroll
      for (int r = 0; r < 8; ++r) { s0[r] = 0.f; s1[r] = 0.f; }
      #pragma unroll
      for (int c = 0; c < 4; ++c) {
        v16bf b0, b1;
        const __bf16* k0p = &Ksh[laneN][c * 32 + hi * 8];
        const __bf16* k1p = &Ksh[16 + laneN][c * 32 + hi * 8];
        #pragma unroll
        for (int j = 0; j < 16; ++j) { b0[j] = k0p[j < 8 ? j : j + 8]; b1[j] = k1p[j < 8 ? j : j + 8]; }
        s0 = __builtin_amdgcn_wmma_f32_16x16x32_bf16(false, qa[c], false, b0, (short)0, s0, false, false);
        s1 = __builtin_amdgcn_wmma_f32_16x16x32_bf16(false, qa[c], false, b1, (short)0, s1, false, false);
      }
      float p0[8], p1[8];
      #pragma unroll
      for (int r = 0; r < 8; ++r) {
        const int qrow = q0 + r + hi * 8;
        float x0 = tanhf(s0[r] * (1.f / SOFTCAP_)) * SOFTCAP_;
        float x1 = tanhf(s1[r] * (1.f / SOFTCAP_)) * SOFTCAP_;
        if (j0 + laneN > qrow)      x0 = -1e9f;
        if (j0 + 16 + laneN > qrow) x1 = -1e9f;
        const float mt   = wave_max16(fmaxf(x0, x1));
        const float mnew = fmaxf(mrow[r], mt);
        const float resc = __expf(mrow[r] - mnew);
        const float e0 = __expf(x0 - mnew);
        const float e1 = __expf(x1 - mnew);
        const float ls = wave_sum16(e0 + e1);
        lrow[r] = lrow[r] * resc + ls;
        mrow[r] = mnew;
        #pragma unroll
        for (int ch = 0; ch < 8; ++ch) accv[ch][r] *= resc;
        p0[r] = e0; p1[r] = e1;
      }
      __bf16* pp = &Psh[wid][0][0];
      #pragma unroll
      for (int r = 0; r < 8; ++r) {
        pp[(r + hi * 8) * 32 + laneN]      = (__bf16)p0[r];
        pp[(r + hi * 8) * 32 + 16 + laneN] = (__bf16)p1[r];
      }
      __builtin_amdgcn_wave_barrier();
      asm volatile("s_wait_dscnt 0" ::: "memory");
      v16bf pa;
      {
        const __bf16* pr = pp + laneN * 32 + hi * 8;
        #pragma unroll
        for (int j = 0; j < 8; ++j) { pa[j] = pr[j]; pa[j + 8] = pr[16 + j]; }
      }
      #pragma unroll
      for (int ch = 0; ch < 8; ++ch) {
        v16bf bv;
        const __bf16* vp = &Vsh[ch * 16 + laneN][hi * 8];
        #pragma unroll
        for (int j = 0; j < 16; ++j) bv[j] = vp[j < 8 ? j : j + 8];
        accv[ch] = __builtin_amdgcn_wmma_f32_16x16x32_bf16(false, pa, false, bv, (short)0, accv[ch], false, false);
      }
    }
    __syncthreads();
  }
  #pragma unroll
  for (int r = 0; r < 8; ++r) {
    const int row  = q0 + r + hi * 8;
    const float invl = 1.f / lrow[r];
    float* op = attn + ((size_t)(b * S_ + row)) * QDIM_ + h * HD_;
    #pragma unroll
    for (int ch = 0; ch < 8; ++ch) op[ch * 16 + laneN] = accv[ch][r] * invl;
  }
}

extern "C" void kernel_launch(void* const* d_in, const int* in_sizes, int n_in,
                              void* d_out, int out_size, void* d_ws, size_t ws_size,
                              hipStream_t stream)
{
  (void)in_sizes; (void)n_in; (void)out_size; (void)ws_size;
  const float* hidden = (const float*)d_in[0];
  const float* freqs  = (const float*)d_in[1];
  const float* w_qkv  = (const float*)d_in[3];
  const float* w_o    = (const float*)d_in[4];
  const float* w_gu   = (const float*)d_in[5];
  const float* w_dn   = (const float*)d_in[6];
  const float* ln1    = (const float*)d_in[7];
  const float* ln2    = (const float*)d_in[8];
  float* out = (float*)d_out;
  char*  ws  = (char*)d_ws;

  constexpr size_t OFF_ACT8 = 0;
  constexpr size_t OFF_SCL  = OFF_ACT8 + (64ull << 20);
  constexpr size_t OFF_SW   = OFF_SCL  + (1ull  << 20);
  constexpr size_t OFF_QW   = OFF_SW   + (1ull  << 20);
  constexpr size_t OFF_BIG  = OFF_QW   + (120ull << 20);
  constexpr size_t OFF_QT   = OFF_BIG  + (480ull << 20);
  constexpr size_t OFF_KT   = OFF_QT   + (64ull  << 20);
  constexpr size_t OFF_VT   = OFF_KT   + (16ull  << 20);
  constexpr size_t OFF_X1   = OFF_VT   + (16ull  << 20);

  int8_t* act8 = (int8_t*)(ws + OFF_ACT8);
  float*  scl  = (float*)(ws + OFF_SCL);
  float*  sw   = (float*)(ws + OFF_SW);
  int8_t* qw   = (int8_t*)(ws + OFF_QW);
  float*  big  = (float*)(ws + OFF_BIG);
  float*  qTb  = (float*)(ws + OFF_QT);
  float*  kTb  = (float*)(ws + OFF_KT);
  float*  vTb  = (float*)(ws + OFF_VT);
  float*  x1   = (float*)(ws + OFF_X1);

  rmsnorm_quant_kernel<<<T_, 256, 0, stream>>>(hidden, ln1, act8, scl, D_);
  quant_rows_kernel<<<QKV_, 256, 0, stream>>>(w_qkv, qw, sw, D_);
  gemm_i8_kernel<<<dim3(QKV_ / 64, T_ / 128), 256, 0, stream>>>(
      act8, scl, qw, sw, nullptr, big, T_, QKV_, D_);
  rope_split_kernel<<<T_, 256, 0, stream>>>(big, freqs, qTb, kTb, vTb);
  attention_kernel<<<dim3(S_ / 128, H_, BSZ_), 256, 0, stream>>>(qTb, kTb, vTb, big);
  quant_rows_kernel<<<T_, 256, 0, stream>>>(big, act8, scl, QDIM_);
  quant_rows_kernel<<<D_, 256, 0, stream>>>(w_o, qw, sw, QDIM_);
  gemm_i8_kernel<<<dim3(D_ / 64, T_ / 128), 256, 0, stream>>>(
      act8, scl, qw, sw, hidden, x1, T_, D_, QDIM_);
  rmsnorm_quant_kernel<<<T_, 256, 0, stream>>>(x1, ln2, act8, scl, D_);
  quant_rows_kernel<<<2 * I_, 256, 0, stream>>>(w_gu, qw, sw, D_);
  gemm_i8_kernel<<<dim3(2 * I_ / 64, T_ / 128), 256, 0, stream>>>(
      act8, scl, qw, sw, nullptr, big, T_, 2 * I_, D_);
  silu_quant_kernel<<<T_, 256, 0, stream>>>(big, act8, scl);
  quant_rows_kernel<<<D_, 256, 0, stream>>>(w_dn, qw, sw, I_);
  gemm_i8_kernel<<<dim3(D_ / 64, T_ / 128), 256, 0, stream>>>(
      act8, scl, qw, sw, x1, out, T_, D_, I_);
}
